// AutoregressiveDecoder_29532195127338
// MI455X (gfx1250) — matrix operand order repaired and checked by reading, not compiled, not run on hardware
//
#include <hip/hip_runtime.h>
#include <hip/hip_bf16.h>
#include <math.h>

typedef __bf16 bf16_t;
typedef __attribute__((ext_vector_type(8)))  float  v8f;
typedef __attribute__((ext_vector_type(4)))  float  v4f;
typedef __attribute__((ext_vector_type(16))) __bf16 v16bf;
typedef __attribute__((ext_vector_type(8)))  __bf16 v8bf;

#define NB   128
#define NT   1024
#define NZ   128
#define NH   512
#define NF   128
#define NKF  2
#define G3   1536   // 3*H

// LDS strides (elements), padded for bank-conflict-free b128 loads
#define HS_BF 520
#define HS_F  516
#define XS_BF 136

// ---- workspace layout (bytes) ----
#define WS_W0X   ((size_t)0)                              // 4*1536*32 bf16
#define WS_WH0   (WS_W0X  + (size_t)4*G3*32*2)            // 16*1536*32 bf16
#define WS_WI1   (WS_WH0  + (size_t)16*G3*32*2)
#define WS_WH1   (WS_WI1  + (size_t)16*G3*32*2)
#define WS_WOUT  (WS_WH1  + (size_t)16*G3*32*2)           // 16*512*32 bf16
#define WS_C0    (WS_WOUT + (size_t)16*512*32*2)          // 8*96*32*8 f32
#define WS_HB    (WS_C0   + (size_t)8*96*32*8*4)          // 512 f32
#define WS_H0    (WS_HB   + (size_t)512*4)                // 128*512 f32
#define WS_H1    (WS_H0   + (size_t)NB*NH*4)

// ------------------------------------------------------------------
// helpers
// ------------------------------------------------------------------
__device__ __forceinline__ v8f wmma_bf(v16bf a, v16bf b, v8f c) {
  return __builtin_amdgcn_wmma_f32_16x16x32_bf16(false, a, false, b, (short)0, c, false, false);
}

__device__ __forceinline__ v16bf load_afrag(const bf16_t* p) {
  v8bf lo = *(const v8bf*)(p);
  v8bf hi = *(const v8bf*)(p + 16);
  return __builtin_shufflevector(lo, hi, 0,1,2,3,4,5,6,7,8,9,10,11,12,13,14,15);
}

// swizzled weight: [kt][n][32] bf16, lane reads k-sub-ranges half*8..half*8+7 and 16+half*8..16+half*8+7
__device__ __forceinline__ v16bf load_bfrag(const bf16_t* w, int kt, int N, int nt,
                                            int m16, int half) {
  const bf16_t* p = w + ((size_t)(kt*N + nt*16 + m16))*32 + half*8;
  v8bf lo = *(const v8bf*)(p);
  v8bf hi = *(const v8bf*)(p + 16);
  return __builtin_shufflevector(lo, hi, 0,1,2,3,4,5,6,7,8,9,10,11,12,13,14,15);
}

__device__ __forceinline__ v8f load_c0frag(const float* c0, int bt, int nt, int lane) {
  const float* p = c0 + (((size_t)bt*96 + nt)*32 + lane)*8;
  v4f a = *(const v4f*)(p);
  v4f b = *(const v4f*)(p + 4);
  return __builtin_shufflevector(a, b, 0,1,2,3,4,5,6,7);
}

__device__ __forceinline__ v8f splat8(float x) {
  v8f r;
#pragma unroll
  for (int i = 0; i < 8; ++i) r[i] = x;
  return r;
}

__device__ __forceinline__ float sigmoidf(float x) { return 1.0f/(1.0f + expf(-x)); }

// one K-accumulation pass over all three gate slices owned by this wave
template<int KT>
__device__ __forceinline__ void gemm_pass(const bf16_t* abuf, int astride,
                                          const bf16_t* wmat,
                                          v8f* aR, v8f* aZ, v8f* aN,
                                          int w, int m16, int half) {
  const bf16_t* abase = abuf + m16*astride + half*8;
  for (int kt = 0; kt < KT; ++kt) {
    v16bf a = load_afrag(abase + kt*32);
#pragma unroll
    for (int j = 0; j < 4; ++j) {
      const int ct = w*4 + j;
      aR[j] = wmma_bf(a, load_bfrag(wmat, kt, G3, ct,      m16, half), aR[j]);
      aZ[j] = wmma_bf(a, load_bfrag(wmat, kt, G3, 32 + ct, m16, half), aZ[j]);
      aN[j] = wmma_bf(a, load_bfrag(wmat, kt, G3, 64 + ct, m16, half), aN[j]);
    }
  }
}

// gates -> h_new; result left in aR
__device__ __forceinline__ void gru_gates(v8f* aR, const v8f* aZ, const v8f* aNi, const v8f* aNh,
                                          const float* hf, int w, int m16, int half) {
#pragma unroll
  for (int j = 0; j < 4; ++j) {
    const int colb = (w*4 + j)*16 + m16;
#pragma unroll
    for (int v = 0; v < 8; ++v) {
      const int mr = v + half*8;
      float r  = sigmoidf(aR[j][v]);
      float u  = sigmoidf(aZ[j][v]);
      float n  = tanhf(aNi[j][v] + r*aNh[j][v]);
      float ho = hf[mr*HS_F + colb];
      aR[j][v] = (1.0f - u)*n + u*ho;
    }
  }
}

__device__ __forceinline__ void h_writeback(const v8f* hn, float* hf, bf16_t* hb,
                                            int w, int m16, int half) {
#pragma unroll
  for (int j = 0; j < 4; ++j) {
    const int colb = (w*4 + j)*16 + m16;
#pragma unroll
    for (int v = 0; v < 8; ++v) {
      const int mr = v + half*8;
      float hv = hn[j][v];
      hf[mr*HS_F + colb]  = hv;
      hb[mr*HS_BF + colb] = (bf16_t)hv;
    }
  }
}

// ------------------------------------------------------------------
// prep kernels
// ------------------------------------------------------------------
__global__ void swizzle_w(bf16_t* dst, int dstN, int colOff,
                          const float* src, int srcN, int srcStride, int kOff, int KT) {
  int idx = blockIdx.x*256 + threadIdx.x;
  int total = srcN*KT*32;
  if (idx >= total) return;
  int j  = idx & 31;
  int r  = idx >> 5;
  int n  = r % srcN;
  int kt = r / srcN;
  dst[((size_t)(kt*dstN + colOff + n))*32 + j] =
      (bf16_t)src[(size_t)n*srcStride + kOff + kt*32 + j];
}

// C0[bt][nt][lane][v] = b_ih0[col] + (col<1024 ? b_hh0[col] : 0) + (z @ Wi0_zpart.T)[b][col]
__global__ void c0_kernel(float* c0, const float* z, const float* wih0,
                          const float* bih0, const float* bhh0) {
  int idx = blockIdx.x*256 + threadIdx.x;
  if (idx >= 8*96*32*8) return;
  int v    = idx & 7;
  int r1   = idx >> 3;
  int lane = r1 & 31;
  int r2   = r1 >> 5;
  int nt   = r2 % 96;
  int bt   = r2 / 96;
  int b    = bt*16 + v + ((lane >> 4) << 3);
  int col  = nt*16 + (lane & 15);
  float acc = bih0[col] + (col < 1024 ? bhh0[col] : 0.0f);
  const float* zr = z + (size_t)b*NZ;
  const float* wr = wih0 + (size_t)col*(NF + NZ) + NF;   // z-part columns
  for (int j = 0; j < NZ; ++j) acc += zr[j]*wr[j];
  c0[idx] = acc;
}

__global__ void hinit_kernel(float* h0i, float* h1i, const float* z,
                             const float* fiw, const float* fib) {
  int idx = blockIdx.x*256 + threadIdx.x;
  if (idx >= 2*NB*NH) return;
  int h = idx & (NH - 1);
  int b = (idx >> 9) & (NB - 1);
  int l = idx >> 16;
  int row = l*NH + h;
  float acc = fib[row];
  const float* zr = z + (size_t)b*NZ;
  const float* wr = fiw + (size_t)row*NZ;
  for (int j = 0; j < NZ; ++j) acc += zr[j]*wr[j];
  float* dst = l ? h1i : h0i;
  dst[(size_t)b*NH + h] = tanhf(acc);
}

__global__ void hb_kernel(float* hb, const float* mub, const float* lsb, const float* ldb) {
  int n = blockIdx.x*256 + threadIdx.x;
  if (n >= 512) return;
  float v;
  if (n < 128)      v = mub[n];
  else if (n < 256) v = lsb[n - 128];
  else              v = ldb[n - 256];
  hb[n] = v;
}

// ------------------------------------------------------------------
// persistent scan kernel: 8 blocks (one 16-row batch tile each), 8 waves
// ------------------------------------------------------------------
__launch_bounds__(256, 1)
__global__ void scan_kernel(const float* __restrict__ x,
                            const float* __restrict__ fvec,
                            const float* __restrict__ start_tok,
                            const float* __restrict__ bhh0,
                            const float* __restrict__ bih1,
                            const float* __restrict__ bhh1,
                            const bf16_t* __restrict__ w0x,
                            const bf16_t* __restrict__ wh0,
                            const bf16_t* __restrict__ wi1,
                            const bf16_t* __restrict__ wh1,
                            const bf16_t* __restrict__ wout,
                            const float* __restrict__ c0,
                            const float* __restrict__ hb,
                            const float* __restrict__ h0i,
                            const float* __restrict__ h1i,
                            float* __restrict__ out) {
  __shared__ float  h0f[16*HS_F];
  __shared__ float  h1f[16*HS_F];
  __shared__ float  yb [16*HS_F];
  __shared__ bf16_t h0b[16*HS_BF];
  __shared__ bf16_t h1b[16*HS_BF];
  __shared__ bf16_t xsb[16*XS_BF];

  const int tid  = threadIdx.x;
  const int lane = tid & 31;
  const int w    = tid >> 5;       // wave id 0..7
  const int m16  = lane & 15;
  const int half = lane >> 4;
  const int bt   = blockIdx.x;     // batch tile 0..7
  const int brow = bt*16;

  // load initial hidden states
  for (int i = tid; i < 16*NH; i += 256) {
    int m = i >> 9, c = i & (NH - 1);
    float v0 = h0i[(size_t)(brow + m)*NH + c];
    float v1 = h1i[(size_t)(brow + m)*NH + c];
    h0f[m*HS_F + c] = v0;  h0b[m*HS_BF + c] = (bf16_t)v0;
    h1f[m*HS_F + c] = v1;  h1b[m*HS_BF + c] = (bf16_t)v1;
  }
  __syncthreads();

  float* oc = out;
  float* os = out + (size_t)NB*NT*NF;
  float* ol = out + (size_t)2*NB*NT*NF;

  v8f aR[4], aZ[4], aNi[4], aNh[4];

#pragma unroll 1
  for (int t = 0; t < NT; ++t) {
    // Opaque zero regenerated every iteration: keeps the streamed weight /
    // C0-fragment addresses loop-variant so LICM cannot hoist (and then
    // spill) the loop-invariant global loads out of the scan loop.
    int oz;
    asm volatile("s_mov_b32 %0, 0" : "=s"(oz));
    const bf16_t* w0x_v  = w0x  + oz;
    const bf16_t* wh0_v  = wh0  + oz;
    const bf16_t* wi1_v  = wi1  + oz;
    const bf16_t* wh1_v  = wh1  + oz;
    const bf16_t* wout_v = wout + oz;
    const float*  c0_v   = c0   + oz;

    // ---- stage x_prev tile (bf16) ----
    for (int i = tid; i < 16*NF; i += 256) {
      int m = i >> 7, c = i & (NF - 1);
      float xv = (t == 0) ? start_tok[c]
                          : x[((size_t)(brow + m)*NT + (t - 1))*NF + c];
      xsb[m*XS_BF + c] = (bf16_t)xv;
    }
    __syncthreads();

    // ---- layer 0 ----
#pragma unroll
    for (int j = 0; j < 4; ++j) {
      const int ct = w*4 + j;
      aR[j]  = load_c0frag(c0_v, bt, ct,      lane);
      aZ[j]  = load_c0frag(c0_v, bt, 32 + ct, lane);
      aNi[j] = load_c0frag(c0_v, bt, 64 + ct, lane);
      aNh[j] = splat8(bhh0[1024 + ct*16 + m16]);
    }
    gemm_pass<4> (xsb, XS_BF, w0x_v, aR, aZ, aNi, w, m16, half);   // input part (K=128)
    gemm_pass<16>(h0b, HS_BF, wh0_v, aR, aZ, aNh, w, m16, half);   // hidden part (K=512)
    gru_gates(aR, aZ, aNi, aNh, h0f, w, m16, half);
    __syncthreads();
    h_writeback(aR, h0f, h0b, w, m16, half);
    __syncthreads();

    // ---- layer 1 ----
#pragma unroll
    for (int j = 0; j < 4; ++j) {
      const int cb = (w*4 + j)*16 + m16;
      aR[j]  = splat8(bih1[cb]        + bhh1[cb]);
      aZ[j]  = splat8(bih1[512 + cb]  + bhh1[512 + cb]);
      aNi[j] = splat8(bih1[1024 + cb]);
      aNh[j] = splat8(bhh1[1024 + cb]);
    }
    gemm_pass<16>(h0b, HS_BF, wi1_v, aR, aZ, aNi, w, m16, half);   // input = h0_new
    gemm_pass<16>(h1b, HS_BF, wh1_v, aR, aZ, aNh, w, m16, half);   // hidden
    gru_gates(aR, aZ, aNi, aNh, h1f, w, m16, half);
    __syncthreads();
    h_writeback(aR, h1f, h1b, w, m16, half);
    __syncthreads();

    // ---- output heads: Y(16x512) = h1_new @ [mu|ls|load]^T + b ----
    {
      v8f ya[4];
#pragma unroll
      for (int j = 0; j < 4; ++j) ya[j] = splat8(hb[(w*4 + j)*16 + m16]);
      const bf16_t* abase = h1b + m16*HS_BF + half*8;
      for (int kt = 0; kt < 16; ++kt) {
        v16bf a = load_afrag(abase + kt*32);
#pragma unroll
        for (int j = 0; j < 4; ++j)
          ya[j] = wmma_bf(a, load_bfrag(wout_v, kt, 512, w*4 + j, m16, half), ya[j]);
      }
#pragma unroll
      for (int j = 0; j < 4; ++j) {
        const int colb = (w*4 + j)*16 + m16;
#pragma unroll
        for (int v = 0; v < 8; ++v) yb[(v + half*8)*HS_F + colb] = ya[j][v];
      }
    }
    __syncthreads();

    // ---- elementwise outputs ----
    {
      const int m = tid >> 4;
      const int q = tid & 15;
      const int b = brow + m;
      const size_t bto = (size_t)b*NT + t;
      const float fv0 = fvec[bto*NKF + 0];
      const float fv1 = fvec[bto*NKF + 1];
#pragma unroll
      for (int i = 0; i < 8; ++i) {
        const int fc = q*8 + i;
        float mu = yb[m*HS_F + fc];
        float ls = yb[m*HS_F + 128 + fc];
        float l0 = yb[m*HS_F + 256 + 2*fc];
        float l1 = yb[m*HS_F + 256 + 2*fc + 1];
        ls = fminf(fmaxf(ls, -5.0f), 2.0f);
        const size_t o = bto*NF + fc;
        oc[o]     = mu + l0*fv0 + l1*fv1;
        os[o]     = expf(ls);
        ol[o*2]   = l0;
        ol[o*2+1] = l1;
      }
    }
    // no barrier needed: next step's LDS writes are fenced by its own barriers
  }
}

// ------------------------------------------------------------------
// host entry
// ------------------------------------------------------------------
extern "C" void kernel_launch(void* const* d_in, const int* in_sizes, int n_in,
                              void* d_out, int out_size, void* d_ws, size_t ws_size,
                              hipStream_t stream) {
  const float* z    = (const float*)d_in[0];
  const float* x    = (const float*)d_in[1];
  const float* f    = (const float*)d_in[2];
  const float* st   = (const float*)d_in[3];
  const float* fiw  = (const float*)d_in[4];
  const float* fib  = (const float*)d_in[5];
  const float* wih0 = (const float*)d_in[6];
  const float* whh0 = (const float*)d_in[7];
  const float* bih0 = (const float*)d_in[8];
  const float* bhh0 = (const float*)d_in[9];
  const float* wih1 = (const float*)d_in[10];
  const float* whh1 = (const float*)d_in[11];
  const float* bih1 = (const float*)d_in[12];
  const float* bhh1 = (const float*)d_in[13];
  const float* muw  = (const float*)d_in[14];
  const float* mub  = (const float*)d_in[15];
  const float* lsw  = (const float*)d_in[16];
  const float* lsb  = (const float*)d_in[17];
  const float* ldw  = (const float*)d_in[18];
  const float* ldb  = (const float*)d_in[19];

  char* ws = (char*)d_ws;
  bf16_t* w0x  = (bf16_t*)(ws + WS_W0X);
  bf16_t* wh0  = (bf16_t*)(ws + WS_WH0);
  bf16_t* wi1  = (bf16_t*)(ws + WS_WI1);
  bf16_t* wh1  = (bf16_t*)(ws + WS_WH1);
  bf16_t* wout = (bf16_t*)(ws + WS_WOUT);
  float*  c0   = (float*)(ws + WS_C0);
  float*  hb   = (float*)(ws + WS_HB);
  float*  h0i  = (float*)(ws + WS_H0);
  float*  h1i  = (float*)(ws + WS_H1);
  float*  out  = (float*)d_out;

  auto sw = [&](bf16_t* dst, int dstN, int colOff, const float* src,
                int srcN, int srcStride, int kOff, int KT) {
    int total = srcN*KT*32;
    swizzle_w<<<(total + 255)/256, 256, 0, stream>>>(dst, dstN, colOff, src,
                                                     srcN, srcStride, kOff, KT);
  };
  sw(w0x,  G3,  0,   wih0, G3,  NF + NZ, 0, 4);   // x-part of W_ih0
  sw(wh0,  G3,  0,   whh0, G3,  NH,      0, 16);
  sw(wi1,  G3,  0,   wih1, G3,  NH,      0, 16);
  sw(wh1,  G3,  0,   whh1, G3,  NH,      0, 16);
  sw(wout, 512, 0,   muw,  128, NH,      0, 16);
  sw(wout, 512, 128, lsw,  128, NH,      0, 16);
  sw(wout, 512, 256, ldw,  256, NH,      0, 16);

  hb_kernel<<<2, 256, 0, stream>>>(hb, mub, lsb, ldb);
  c0_kernel<<<768, 256, 0, stream>>>(c0, z, wih0, bih0, bhh0);
  hinit_kernel<<<512, 256, 0, stream>>>(h0i, h1i, z, fiw, fib);

  scan_kernel<<<8, 256, 0, stream>>>(x, f, st, bhh0, bih1, bhh1,
                                     w0x, wh0, wi1, wh1, wout,
                                     c0, hb, h0i, h1i, out);
}
